// MultiHeadSelfAttention_67061619360019
// MI455X (gfx1250) — matrix operand order repaired and checked by reading, not compiled, not run on hardware
//
#include <hip/hip_runtime.h>

// ---------------- problem constants (from the reference) ----------------
#define NHEAD   16
#define DHEAD   64
#define D_MODEL 1024   // NHEAD * DHEAD
#define BATCH   2
#define SEQ     2048
#define NEG_BIG 1e12f

typedef __attribute__((ext_vector_type(16))) __bf16 v16bf;
typedef __attribute__((ext_vector_type(8)))  __bf16 v8bf;
typedef __attribute__((ext_vector_type(8)))  float  v8f;

__device__ __forceinline__ __bf16 f2bf(float x) { return (__bf16)x; }

__device__ __forceinline__ v16bf cat8(v8bf lo, v8bf hi) {
    return __builtin_shufflevector(lo, hi, 0, 1, 2, 3, 4, 5, 6, 7,
                                           8, 9, 10, 11, 12, 13, 14, 15);
}

// ---------------- Tensor Data Mover (guarded, with sync fallback) -------
#if defined(__gfx1250__) && __has_builtin(__builtin_amdgcn_tensor_load_to_lds)
#define HAVE_TDM 1
#else
#define HAVE_TDM 0
#endif
#if __has_include(<hip/amd_detail/amd_gfx1250_TDM.h>)
#define TDM_SIX_ARGS 1
#endif

#if HAVE_TDM
typedef unsigned int u32x4 __attribute__((ext_vector_type(4)));
typedef int          i32x4 __attribute__((ext_vector_type(4)));
typedef int          i32x8 __attribute__((ext_vector_type(8)));

// Linear copy of n_elems bf16 elements (global -> LDS) through the TDM.
// D# packed per CDNA5 ISA 8.3/8.4: 1-D tile (tile_dim0=n, tile_dim1=1).
__device__ __forceinline__ void tdm_load_lin(unsigned lds_byte_off,
                                             const void* gsrc,
                                             unsigned n_elems) {
    const unsigned long long ga = (unsigned long long)(size_t)gsrc;
    u32x4 g0;
    g0[0] = 1u;                                    // count=1 (valid user D#)
    g0[1] = lds_byte_off;                          // lds_addr
    g0[2] = (unsigned)(ga & 0xFFFFFFFFu);          // global_addr[31:0]
    g0[3] = (unsigned)((ga >> 32) & 0x1FFFFFFu)    // global_addr[56:32]
          | (2u << 30);                            // type = 2 ("image")
    i32x8 g1;
    g1[0] = (int)(1u << 16);                       // data_size=1 -> 2 bytes
    g1[1] = (int)((n_elems & 0xFFFFu) << 16);      // tensor_dim0[15:0]
    g1[2] = (int)(((n_elems >> 16) & 0xFFFFu)      // tensor_dim0[31:16]
          | (1u << 16));                           // tensor_dim1[15:0] = 1
    g1[3] = (int)((n_elems & 0xFFFFu) << 16);      // tile_dim0 = n_elems
    g1[4] = 1;                                     // tile_dim1=1, tile_dim2=0
    g1[5] = (int)n_elems;                          // tensor_dim0_stride[31:0]
    g1[6] = 0;                                     // stride0[47:32]|stride1[15:0]
    g1[7] = 0;                                     // stride1[47:16]
    const i32x4 z4 = {0, 0, 0, 0};
#if defined(TDM_SIX_ARGS)
    const i32x8 z8 = {0, 0, 0, 0, 0, 0, 0, 0};
    __builtin_amdgcn_tensor_load_to_lds(g0, g1, z4, z4, z8, 0);
#else
    __builtin_amdgcn_tensor_load_to_lds(g0, g1, z4, z4, 0);
#endif
}

__device__ __forceinline__ void tdm_wait0() {
#if __has_builtin(__builtin_amdgcn_s_wait_tensorcnt)
    __builtin_amdgcn_s_wait_tensorcnt(0);
#else
    asm volatile("s_wait_tensorcnt 0x0" ::: "memory");
#endif
}
#endif  // HAVE_TDM

// =======================================================================
// Kernel: flash attention per (b, h, 128-query tile).   (placed first so
// the disasm snippet shows its TDM stage-in path)
// =======================================================================
__global__ __launch_bounds__(256) void attn_kernel(
    const __bf16* __restrict__ qb, const __bf16* __restrict__ kbuf,
    const __bf16* __restrict__ vbuf,
    const int* __restrict__ q_len, const int* __restrict__ v_len,
    float* __restrict__ out)
{
    __shared__ __bf16 Ks[32][DHEAD];     // [key][d]  (B source for scores)
    __shared__ __bf16 Vs[32][DHEAD];     // raw V tile
    __shared__ __bf16 Vt[DHEAD][32];     // V^T [d][key] (B source for PV)
    __shared__ float  Sbuf[8][16][32];   // per-wave scores
    __shared__ __bf16 Pbuf[8][16][32];   // per-wave probabilities
    __shared__ float  row_m[8][16];
    __shared__ float  row_l[8][16];
    __shared__ float  row_scale[8][16];
    __shared__ float  row_mult[8][16];

    const int b  = blockIdx.z;
    const int h  = blockIdx.y;
    const int q0 = blockIdx.x * 128;

    const int tid  = threadIdx.x;
    const int lane = tid & 31;
    const int wave = tid >> 5;
    const int hi   = lane >> 4;
    const int nn   = lane & 15;

    const size_t head_base = ((size_t)(b * NHEAD + h)) * SEQ * DHEAD;

    // ---- Q fragments: loop-invariant, straight from global to registers ----
    v16bf qa[2];
    {
        const __bf16* qrow =
            qb + head_base + (size_t)(q0 + wave * 16 + nn) * DHEAD;
        #pragma unroll
        for (int kk2 = 0; kk2 < 2; ++kk2)
            qa[kk2] = cat8(
                *reinterpret_cast<const v8bf*>(qrow + kk2 * 32 + 16 * hi),
                *reinterpret_cast<const v8bf*>(qrow + kk2 * 32 + 16 * hi + 8));
    }
    if (lane < 16) {
        row_m[wave][lane] = -1e30f;
        row_l[wave][lane] = 0.0f;
    }
    __syncthreads();

    const int vlen = v_len[b];
    const int qlen = q_len[b];
    const int nkb  = (vlen + 31) >> 5;

    v8f Oacc[4] = {};

    for (int kbI = 0; kbI < nkb; ++kbI) {
        const __bf16* ksrc = kbuf + head_base + (size_t)(kbI * 32) * DHEAD;
        const __bf16* vsrc = vbuf + head_base + (size_t)(kbI * 32) * DHEAD;

#if HAVE_TDM
        if (wave == 0) {  // one wave drives the DMA; EXEC-independent issue
            tdm_load_lin((unsigned)(size_t)&Ks[0][0], ksrc, 32 * DHEAD);
            tdm_load_lin((unsigned)(size_t)&Vs[0][0], vsrc, 32 * DHEAD);
            tdm_wait0();
        }
#else
        reinterpret_cast<uint4*>(&Ks[0][0])[tid] =
            reinterpret_cast<const uint4*>(ksrc)[tid];
        reinterpret_cast<uint4*>(&Vs[0][0])[tid] =
            reinterpret_cast<const uint4*>(vsrc)[tid];
#endif
        if (wave == 1 && kbI + 1 < nkb) {  // global_prefetch_b8 of next tiles
            __builtin_prefetch(ksrc + 32 * DHEAD + lane * 64, 0, 1);
            __builtin_prefetch(vsrc + 32 * DHEAD + lane * 64, 0, 1);
        }
        __syncthreads();

        // transpose V into Vt[d][key] so PV B-fragments are contiguous
        {
            const int key = tid >> 3;
            const int d0  = (tid & 7) * 8;
            const v8bf v = *reinterpret_cast<const v8bf*>(&Vs[key][d0]);
            #pragma unroll
            for (int j = 0; j < 8; ++j) Vt[d0 + j][key] = v[j];
        }
        __syncthreads();

        // ---- scores: S(16x32) = Q . K^T, two 16x16 tiles, K=64 ----
        #pragma unroll
        for (int tj = 0; tj < 2; ++tj) {
            v8f s = {};
            #pragma unroll
            for (int kk2 = 0; kk2 < 2; ++kk2) {
                const v16bf bfr = *reinterpret_cast<const v16bf*>(
                    &Ks[tj * 16 + nn][kk2 * 32 + 16 * hi]);
                s = __builtin_amdgcn_wmma_f32_16x16x32_bf16(
                    false, qa[kk2], false, bfr, (short)0, s, false, false);
            }
            #pragma unroll
            for (int r = 0; r < 8; ++r)
                Sbuf[wave][r + 8 * hi][tj * 16 + nn] = s[r] * 0.125f;
        }
        __syncthreads();

        // ---- online softmax: all 32 lanes, each owns half a row ----
        {
            const int   row   = nn;
            const float m_old = row_m[wave][row];
            float s16[16];
            float mh = -1e30f;
            #pragma unroll
            for (int q2 = 0; q2 < 4; ++q2) {
                const float4 s4 = *reinterpret_cast<const float4*>(
                    &Sbuf[wave][row][16 * hi + 4 * q2]);
                const float t4[4] = {s4.x, s4.y, s4.z, s4.w};
                #pragma unroll
                for (int c = 0; c < 4; ++c) {
                    float sv = t4[c];
                    if (kbI * 32 + 16 * hi + 4 * q2 + c >= vlen) sv -= NEG_BIG;
                    s16[q2 * 4 + c] = sv;
                    mh = fmaxf(mh, sv);
                }
            }
            mh = fmaxf(mh, __shfl_xor(mh, 16));
            const float m_new = fmaxf(m_old, mh);
            float sum = 0.0f;
            v16bf pv;
            #pragma unroll
            for (int j = 0; j < 16; ++j) {
                const float p = __expf(s16[j] - m_new);
                sum += p;
                pv[j] = f2bf(p);
            }
            *reinterpret_cast<v16bf*>(&Pbuf[wave][row][16 * hi]) = pv;
            sum += __shfl_xor(sum, 16);
            if (hi == 0) {
                const float scale = __expf(m_old - m_new);
                row_scale[wave][row] = scale;
                row_m[wave][row]     = m_new;
                row_l[wave][row]     = row_l[wave][row] * scale + sum;
            }
        }
        __syncthreads();

        // ---- O = O*scale + P(16x32) @ V(32x64), 4 WMMAs ----
        const __bf16* prow = &Pbuf[wave][nn][0];
        const v16bf pa = cat8(*reinterpret_cast<const v8bf*>(prow + 16 * hi),
                              *reinterpret_cast<const v8bf*>(prow + 16 * hi + 8));
        #pragma unroll
        for (int nt = 0; nt < 4; ++nt) {
            #pragma unroll
            for (int r = 0; r < 8; ++r)
                Oacc[nt][r] *= row_scale[wave][r + 8 * hi];
            const v16bf bfr =
                *reinterpret_cast<const v16bf*>(&Vt[nt * 16 + nn][16 * hi]);
            Oacc[nt] = __builtin_amdgcn_wmma_f32_16x16x32_bf16(
                false, pa, false, bfr, (short)0, Oacc[nt], false, false);
        }
        __syncthreads();   // protect Ks/Vs/Vt before next stage-in
    }

    // ---- epilogue: 1/l and multiplicative q-length mask ----
    if (lane < 16) {
        const float l  = row_l[wave][lane];
        const float iv = (l > 0.0f) ? (1.0f / l) : 0.0f;
        const int   qg = q0 + wave * 16 + lane;
        row_mult[wave][lane] = (qg < qlen) ? iv : 0.0f;
    }
    __syncthreads();

    #pragma unroll
    for (int nt = 0; nt < 4; ++nt) {
        #pragma unroll
        for (int r = 0; r < 8; ++r) {
            const int row = r + 8 * hi;
            const int qg  = q0 + wave * 16 + row;
            out[((size_t)(b * SEQ + qg)) * D_MODEL + h * DHEAD + nt * 16 + nn] =
                Oacc[nt][r] * row_mult[wave][row];
        }
    }
}

// =======================================================================
// Kernel: fused QKV projection -> bf16 [B, H, L, DHEAD].
// Block = 256 thr = 8 waves; block tile M=64, N=64, BK=64.
// Each wave: one A fragment reused over two N-tiles -> 4 WMMAs / stage.
// =======================================================================
__global__ __launch_bounds__(256) void qkv_proj_kernel(
    const float* __restrict__ Qseq, const float* __restrict__ Kseq,
    const float* __restrict__ Vseq,
    const float* __restrict__ WQ, const float* __restrict__ WK,
    const float* __restrict__ WV,
    __bf16* __restrict__ qb, __bf16* __restrict__ kb, __bf16* __restrict__ vb)
{
    const int z = blockIdx.z;
    const float* X   = (z == 0) ? Qseq : (z == 1) ? Kseq : Vseq;
    const float* W   = (z == 0) ? WQ   : (z == 1) ? WK   : WV;
    __bf16*      dst = (z == 0) ? qb   : (z == 1) ? kb   : vb;

    __shared__ __bf16 Xs[64][64];   // X rows x K     (A source, row-major)
    __shared__ __bf16 Wt[64][64];   // W^T [col][k]   (B source, N-major)
    __shared__ __bf16 Cs[64][64];   // output staging

    const int tid  = threadIdx.x;
    const int lane = tid & 31;
    const int wave = tid >> 5;
    const int wm   = wave >> 1;    // 0..3 (16-row tile)
    const int wn   = wave & 1;     // 0..1 (32-col strip)
    const int hi   = lane >> 4;
    const int mr   = lane & 15;

    const int row0 = blockIdx.y * 64;
    const int col0 = blockIdx.x * 64;

    v8f acc[2] = {};

    for (int kk = 0; kk < D_MODEL; kk += 64) {
        // stage X chunk 64x64 (4 float4 per thread, packed b64 LDS stores)
        #pragma unroll
        for (int it = 0; it < 4; ++it) {
            const int idx = tid + it * 256;   // 1024 float4 chunks
            const int r   = idx >> 4;
            const int c4  = idx & 15;
            const float4 v = *reinterpret_cast<const float4*>(
                X + (size_t)(row0 + r) * D_MODEL + kk + c4 * 4);
            __bf16 x4[4] = {f2bf(v.x), f2bf(v.y), f2bf(v.z), f2bf(v.w)};
            *reinterpret_cast<uint2*>(&Xs[r][c4 * 4]) =
                *reinterpret_cast<const uint2*>(x4);
        }
        // stage W chunk 64x64 transposed: Wt[col][k]
        #pragma unroll
        for (int it = 0; it < 4; ++it) {
            const int idx = tid + it * 256;
            const int r   = idx >> 4;         // k row 0..63
            const int c4  = idx & 15;         // 4-col group
            const float4 v = *reinterpret_cast<const float4*>(
                W + (size_t)(kk + r) * D_MODEL + col0 + c4 * 4);
            Wt[c4 * 4 + 0][r] = f2bf(v.x);
            Wt[c4 * 4 + 1][r] = f2bf(v.y);
            Wt[c4 * 4 + 2][r] = f2bf(v.z);
            Wt[c4 * 4 + 3][r] = f2bf(v.w);
        }
        __syncthreads();

        #pragma unroll
        for (int kk2 = 0; kk2 < 2; ++kk2) {
            const __bf16* arow = &Xs[wm * 16 + mr][kk2 * 32];
            const v16bf a =
                cat8(*reinterpret_cast<const v8bf*>(arow + 16 * hi),
                     *reinterpret_cast<const v8bf*>(arow + 16 * hi + 8));
            #pragma unroll
            for (int nt2 = 0; nt2 < 2; ++nt2) {
                const v16bf bfr = *reinterpret_cast<const v16bf*>(
                    &Wt[wn * 32 + nt2 * 16 + mr][kk2 * 32 + 16 * hi]);
                acc[nt2] = __builtin_amdgcn_wmma_f32_16x16x32_bf16(
                    false, a, false, bfr, (short)0, acc[nt2], false, false);
            }
        }
        __syncthreads();
    }

    // C tiles -> LDS bf16 staging, then coalesced b128 stores
    #pragma unroll
    for (int nt2 = 0; nt2 < 2; ++nt2)
        #pragma unroll
        for (int r = 0; r < 8; ++r)
            Cs[wm * 16 + r + 8 * hi][wn * 32 + nt2 * 16 + mr] =
                f2bf(acc[nt2][r]);
    __syncthreads();
    #pragma unroll
    for (int it = 0; it < 2; ++it) {
        const int idx  = tid + it * 256;      // 512 uint4 chunks
        const int row  = idx >> 3;            // 0..63
        const int c0   = (idx & 7) * 8;       // 0..56
        const int grow = row0 + row;
        const int gcol = col0 + c0;
        const int b = grow >> 11;
        const int l = grow & (SEQ - 1);
        const int h = gcol >> 6;
        const int d = gcol & (DHEAD - 1);
        *reinterpret_cast<uint4*>(
            dst + (((size_t)(b * NHEAD + h)) * SEQ + l) * DHEAD + d) =
            *reinterpret_cast<const uint4*>(&Cs[row][c0]);
    }
}

// =======================================================================
extern "C" void kernel_launch(void* const* d_in, const int* in_sizes, int n_in,
                              void* d_out, int out_size, void* d_ws, size_t ws_size,
                              hipStream_t stream) {
    (void)in_sizes; (void)n_in; (void)out_size; (void)ws_size;
    const float* Qseq = (const float*)d_in[0];
    const float* Kseq = (const float*)d_in[1];
    const float* Vseq = (const float*)d_in[2];
    const int*   qlen = (const int*)d_in[3];
    const int*   vlen = (const int*)d_in[4];
    const float* WQ   = (const float*)d_in[5];
    const float* WK   = (const float*)d_in[6];
    const float* WV   = (const float*)d_in[7];
    float* out = (float*)d_out;

    const size_t per = (size_t)BATCH * NHEAD * SEQ * DHEAD;
    __bf16* qb = (__bf16*)d_ws;
    __bf16* kb = qb + per;
    __bf16* vb = kb + per;

    qkv_proj_kernel<<<dim3(16, 64, 3), 256, 0, stream>>>(
        Qseq, Kseq, Vseq, WQ, WK, WV, qb, kb, vb);

    attn_kernel<<<dim3(16, NHEAD, BATCH), 256, 0, stream>>>(
        qb, kb, vb, qlen, vlen, out);
}
